// LSNNCell_12575664243448
// MI455X (gfx1250) — matrix operand order repaired and checked by reading, not compiled, not run on hardware
//
#include <hip/hip_runtime.h>

typedef __attribute__((ext_vector_type(16))) __bf16 v16bf;
typedef __attribute__((ext_vector_type(8)))  float  v8f;

#define BATCH 4096
#define N_IN  1024
#define UNITS 2048
#define N_OUT 1024

#define C_THR    0.4f
#define C_INVTHR 2.5f
#define C_NREF   5.0f
#define C_DECAY  0.9512294245007140f   // exp(-1/20)
#define C_KAPPA  0.9512294245007140f   // exp(-1/20)
#define C_DECAYB 0.9983347214509174f   // exp(-1/600)
#define C_BETA   1.6f

// ---------------------------------------------------------------------------
// Pack f32 row-major [M,K] into WMMA A-fragments (16x32 bf16 tiles).
// tile_id = mt*(K/32) + kt ; within tile: element index = lane*16 + e.
// lane: row = mt*16 + (lane&15), h = lane>>4
// e:    k   = kt*32 + 16*h + e     (contiguous k order, matches B packing)
// s = log2(K/32)  (K = 32<<s)
// ---------------------------------------------------------------------------
__global__ __launch_bounds__(256)
void pack_a_kernel(const float* __restrict__ src, __bf16* __restrict__ dst, int s)
{
    int t    = blockIdx.x * 256 + threadIdx.x;
    int tile = t >> 9;
    int u    = t & 511;
    int lane = u >> 4;
    int e    = u & 15;
    int mt   = tile >> s;
    int kt   = tile & ((1 << s) - 1);
    int row  = (mt << 4) + (lane & 15);
    int h    = lane >> 4;
    int kk   = 16 * h + e;
    int k    = (kt << 5) + kk;
    dst[t] = (__bf16)src[((size_t)row << (5 + s)) + k];
}

// ---------------------------------------------------------------------------
// Pack f32 row-major [K,N] into WMMA B-fragments (32x16 bf16 tiles).
// tile_id = nt*(K/32) + kt ; lane: col = nt*16 + (lane&15), h = lane>>4
// e: k = kt*32 + 16*h + e                               (contiguous k order, matches A packing)
// s = log2(K/32), nsh = log2(N)
// ---------------------------------------------------------------------------
__global__ __launch_bounds__(256)
void pack_w_kernel(const float* __restrict__ src, __bf16* __restrict__ dst, int s, int nsh)
{
    int t    = blockIdx.x * 256 + threadIdx.x;
    int tile = t >> 9;
    int u    = t & 511;
    int lane = u >> 4;
    int e    = u & 15;
    int nt   = tile >> s;
    int kt   = tile & ((1 << s) - 1);
    int col  = (nt << 4) + (lane & 15);
    int h    = lane >> 4;
    int k    = (kt << 5) + (h << 4) + e;
    dst[t] = (__bf16)src[((size_t)k << nsh) + col];
}

// ---------------------------------------------------------------------------
// One GEMM segment: acc[4][4] += A(4 m-tiles) x B(4 n-tiles) over nKt k-tiles.
// 16 WMMAs per k-chunk against 8 x b128-pair loads.
// ---------------------------------------------------------------------------
__device__ __forceinline__ void mma_block(const __bf16* __restrict__ Abase,
                                          const __bf16* __restrict__ Bbase,
                                          int nKt, int mtile0, int ntile0,
                                          int lane, v8f acc[4][4])
{
    const __bf16* ap[4];
    const __bf16* bp[4];
#pragma unroll
    for (int i = 0; i < 4; ++i)
        ap[i] = Abase + (size_t)(mtile0 + i) * nKt * 512 + lane * 16;
#pragma unroll
    for (int j = 0; j < 4; ++j)
        bp[j] = Bbase + (size_t)(ntile0 + j) * nKt * 512 + lane * 16;

    for (int kt = 0; kt < nKt; ++kt) {
        v16bf fa[4];
#pragma unroll
        for (int i = 0; i < 4; ++i)
            fa[i] = *(const v16bf*)(ap[i] + (size_t)kt * 512);
#pragma unroll
        for (int j = 0; j < 4; ++j) {
            v16bf fb = *(const v16bf*)(bp[j] + (size_t)kt * 512);
#pragma unroll
            for (int i = 0; i < 4; ++i)
                acc[i][j] = __builtin_amdgcn_wmma_f32_16x16x32_bf16(
                    false, fa[i], false, fb, (short)0, acc[i][j], false, false);
        }
    }
}

// ---------------------------------------------------------------------------
// Phase 1: i_t = x@w_in + z@w_rec, fused LSNN state update.
// Each wave: 64x64 output tile (4 M-tiles x 4 N-tiles), K = 1024 + 2048.
// Waves in a workgroup share the same n-block (B-fragment reuse in-cache).
// ---------------------------------------------------------------------------
__global__ __launch_bounds__(256)
void lsnn_phase1(const __bf16* __restrict__ xA, const __bf16* __restrict__ zA,
                 const __bf16* __restrict__ winB, const __bf16* __restrict__ wrecB,
                 const float* __restrict__ v_in, const float* __restrict__ z_in,
                 const float* __restrict__ r_in, const float* __restrict__ b_in,
                 float* __restrict__ o_z, float* __restrict__ o_v,
                 float* __restrict__ o_r, float* __restrict__ o_b)
{
    const int lane = threadIdx.x & 31;
    const int wave = (blockIdx.x * blockDim.x + threadIdx.x) >> 5;
    const int mb   = wave & 63;   // 64 m-blocks * 64 rows = 4096 (fast index)
    const int nb   = wave >> 6;   // 32 n-blocks * 64 cols = 2048

    v8f acc[4][4];
    v8f zero = {};
#pragma unroll
    for (int i = 0; i < 4; ++i)
#pragma unroll
        for (int j = 0; j < 4; ++j) acc[i][j] = zero;

    mma_block(xA, winB, 32, mb * 4, nb * 4, lane, acc);   // x @ w_in   (K=1024)
    mma_block(zA, wrecB, 64, mb * 4, nb * 4, lane, acc);  // z @ w_rec  (K=2048)

    // ---- fused elementwise epilogue ----
    const int h   = lane >> 4;
    const int nlo = lane & 15;
#pragma unroll
    for (int i = 0; i < 4; ++i) {
#pragma unroll
        for (int j = 0; j < 4; ++j) {
            v8f t = acc[i][j];
            int col    = nb * 64 + j * 16 + nlo;
            float beta = (col >= 1024) ? C_BETA : 0.0f;
#pragma unroll
            for (int e = 0; e < 8; ++e) {
                int row    = mb * 64 + i * 16 + 8 * h + e; // C/D layout: VGPR e -> M=e (+8 for lanes 16-31)
                size_t idx = (size_t)row * UNITS + col;
                float it = t[e];
                float zf = z_in[idx];
                float vf = v_in[idx];
                float rf = r_in[idx];
                float bf = b_in[idx];
                float nbv    = C_DECAYB * bf + zf;
                float thr_ac = C_THR + nbv * beta;
                float nv     = C_DECAY * vf + it - zf * C_THR;
                float vs     = (nv - thr_ac) * C_INVTHR;
                float nz     = (rf > 0.0f) ? 0.0f : ((vs > 0.0f) ? 1.0f : 0.0f);
                float nr     = fminf(fmaxf(rf + C_NREF * nz - 1.0f, 0.0f), C_NREF);
                o_z[idx] = nz;
                o_v[idx] = nv;
                o_r[idx] = nr;
                o_b[idx] = nbv;
            }
        }
    }
}

// ---------------------------------------------------------------------------
// Phase 2: new_out = KAPPA*out + new_z @ w_out   (4096 x 2048 x 1024)
// ---------------------------------------------------------------------------
__global__ __launch_bounds__(256)
void lsnn_phase2(const __bf16* __restrict__ nzA, const __bf16* __restrict__ woutB,
                 const float* __restrict__ out_in, float* __restrict__ o_out)
{
    const int lane = threadIdx.x & 31;
    const int wave = (blockIdx.x * blockDim.x + threadIdx.x) >> 5;
    const int mb   = wave & 63;   // 64 m-blocks * 64 rows = 4096
    const int nb   = wave >> 6;   // 16 n-blocks * 64 cols = 1024

    v8f acc[4][4];
    v8f zero = {};
#pragma unroll
    for (int i = 0; i < 4; ++i)
#pragma unroll
        for (int j = 0; j < 4; ++j) acc[i][j] = zero;

    mma_block(nzA, woutB, 64, mb * 4, nb * 4, lane, acc); // new_z @ w_out (K=2048)

    const int h   = lane >> 4;
    const int nlo = lane & 15;
#pragma unroll
    for (int i = 0; i < 4; ++i) {
#pragma unroll
        for (int j = 0; j < 4; ++j) {
            v8f t = acc[i][j];
            int col = nb * 64 + j * 16 + nlo;
#pragma unroll
            for (int e = 0; e < 8; ++e) {
                int row    = mb * 64 + i * 16 + 8 * h + e;
                size_t idx = (size_t)row * N_OUT + col;
                o_out[idx] = C_KAPPA * out_in[idx] + t[e];
            }
        }
    }
}

// ---------------------------------------------------------------------------
extern "C" void kernel_launch(void* const* d_in, const int* in_sizes, int n_in,
                              void* d_out, int out_size, void* d_ws, size_t ws_size,
                              hipStream_t stream)
{
    const float* x     = (const float*)d_in[0];
    const float* v     = (const float*)d_in[1];
    const float* z     = (const float*)d_in[2];
    const float* r     = (const float*)d_in[3];
    const float* outp  = (const float*)d_in[4];
    const float* b     = (const float*)d_in[5];
    const float* w_in  = (const float*)d_in[6];
    const float* w_rec = (const float*)d_in[7];
    const float* w_out = (const float*)d_in[8];

    float* o     = (float*)d_out;
    float* o_out = o;                                   // [4096,1024]
    float* o_z   = o_out + (size_t)BATCH * N_OUT;       // [4096,2048]
    float* o_v   = o_z + (size_t)BATCH * UNITS;
    float* o_r   = o_v + (size_t)BATCH * UNITS;
    float* o_b   = o_r + (size_t)BATCH * UNITS;

    char* ws = (char*)d_ws;                             // 56 MB used
    __bf16* xA    = (__bf16*)(ws);                      //  8 MB
    __bf16* zA    = (__bf16*)(ws + (size_t)( 8u << 20));// 16 MB
    __bf16* nzA   = (__bf16*)(ws + (size_t)(24u << 20));// 16 MB
    __bf16* winB  = (__bf16*)(ws + (size_t)(40u << 20));//  4 MB
    __bf16* wrecB = (__bf16*)(ws + (size_t)(44u << 20));//  8 MB
    __bf16* woutB = (__bf16*)(ws + (size_t)(52u << 20));//  4 MB

    pack_a_kernel<<<(BATCH * N_IN) / 256, 256, 0, stream>>>(x, xA, 5);
    pack_a_kernel<<<(BATCH * UNITS) / 256, 256, 0, stream>>>(z, zA, 6);
    pack_w_kernel<<<(N_IN * UNITS) / 256, 256, 0, stream>>>(w_in, winB, 5, 11);
    pack_w_kernel<<<(UNITS * UNITS) / 256, 256, 0, stream>>>(w_rec, wrecB, 6, 11);
    pack_w_kernel<<<(UNITS * N_OUT) / 256, 256, 0, stream>>>(w_out, woutB, 6, 10);

    // 2048 waves: 64 m-blocks x 32 n-blocks, 8 waves/block (same nb per block)
    lsnn_phase1<<<256, 256, 0, stream>>>(xA, zA, winB, wrecB, v, z, r, b,
                                         o_z, o_v, o_r, o_b);

    pack_a_kernel<<<(BATCH * UNITS) / 256, 256, 0, stream>>>(o_z, nzA, 6);

    // 1024 waves: 64 m-blocks x 16 n-blocks
    lsnn_phase2<<<128, 256, 0, stream>>>(nzA, woutB, outp, o_out);
}
